// FBSBlock_21337397527051
// MI455X (gfx1250) — matrix operand order repaired and checked by reading, not compiled, not run on hardware
//
#include <hip/hip_runtime.h>
#include <hip/hip_bf16.h>

#define BDIM 8
#define SDIM 2048
#define DDIM 1024
#define DCDIM 1024
#define BS (BDIM*SDIM)

typedef __attribute__((ext_vector_type(16))) __bf16 v16bf;
typedef __attribute__((ext_vector_type(8)))  float  v8f;

union Frag { v16bf v; uint4 q[2]; };

static __device__ __forceinline__ unsigned short f2bf(float f) {
  union { float f; unsigned u; } v; v.f = f;
  unsigned r = v.u + 0x7FFFu + ((v.u >> 16) & 1u);   // round-to-nearest-even
  return (unsigned short)(r >> 16);
}

static __device__ __forceinline__ v8f wmma_bf16(Frag a, Frag b, v8f c) {
  return __builtin_amdgcn_wmma_f32_16x16x32_bf16(
      /*neg_a=*/false, a.v, /*neg_b=*/false, b.v,
      /*c_mod=*/(short)0, c, /*reuse_a=*/false, /*reuse_b=*/false);
}

// ---------------------------------------------------------------------------
// K1: per-token label argmax (wave32 reduction) + h -> bf16 conversion
// ---------------------------------------------------------------------------
__global__ __launch_bounds__(256) void k_labels_h2bf(
    const float* __restrict__ h, const float* __restrict__ Wlab,
    const float* __restrict__ blab, int* __restrict__ labels,
    unsigned short* __restrict__ hbf)
{
  int wid = threadIdx.x >> 5, lane = threadIdx.x & 31;
  size_t token = (size_t)blockIdx.x * 8 + wid;
  const float* row = h + token * DDIM;
  unsigned short* orow = hbf + token * DDIM;
  float a0 = 0.f, a1 = 0.f, a2 = 0.f, a3 = 0.f;
#pragma unroll
  for (int it = 0; it < 8; ++it) {
    int d = it * 128 + lane * 4;
    float4 hv = *(const float4*)(row + d);
    ushort4 o;
    o.x = f2bf(hv.x); o.y = f2bf(hv.y); o.z = f2bf(hv.z); o.w = f2bf(hv.w);
    *(ushort4*)(orow + d) = o;
    float4 w0 = *(const float4*)(Wlab + (size_t)(d + 0) * 4);
    float4 w1 = *(const float4*)(Wlab + (size_t)(d + 1) * 4);
    float4 w2 = *(const float4*)(Wlab + (size_t)(d + 2) * 4);
    float4 w3 = *(const float4*)(Wlab + (size_t)(d + 3) * 4);
    a0 += hv.x * w0.x + hv.y * w1.x + hv.z * w2.x + hv.w * w3.x;
    a1 += hv.x * w0.y + hv.y * w1.y + hv.z * w2.y + hv.w * w3.y;
    a2 += hv.x * w0.z + hv.y * w1.z + hv.z * w2.z + hv.w * w3.z;
    a3 += hv.x * w0.w + hv.y * w1.w + hv.z * w2.w + hv.w * w3.w;
  }
#pragma unroll
  for (int off = 16; off; off >>= 1) {
    a0 += __shfl_xor(a0, off, 32);
    a1 += __shfl_xor(a1, off, 32);
    a2 += __shfl_xor(a2, off, 32);
    a3 += __shfl_xor(a3, off, 32);
  }
  if (lane == 0) {
    float l0 = a0 + blab[0], l1 = a1 + blab[1], l2 = a2 + blab[2], l3 = a3 + blab[3];
    int best = 0; float bvv = l0;
    if (l1 > bvv) { bvv = l1; best = 1; }
    if (l2 > bvv) { bvv = l2; best = 2; }
    if (l3 > bvv) { bvv = l3; best = 3; }
    labels[token] = best;
  }
}

// ---------------------------------------------------------------------------
// K2: sequential BIOS chunk scan per batch row (chunks = contiguous ranges)
// ---------------------------------------------------------------------------
__global__ void k_chunk_scan(const int* __restrict__ labels,
                             int* __restrict__ cstart, int* __restrict__ clen,
                             int* __restrict__ nck)
{
  int b = threadIdx.x;
  if (b >= BDIM) return;
  const int* lab = labels + (size_t)b * SDIM;
  int* cs = cstart + (size_t)b * SDIM;
  int* cl = clen   + (size_t)b * SDIM;
  int nc = 0; bool open = false;
  for (int s = 0; s < SDIM; ++s) {
    int L = lab[s];
    bool cont = (L == 1) && open;
    if (!cont) { cs[nc] = s; cl[nc] = 1; nc++; }
    else       { cl[nc - 1]++; }
    open = (L == 0) || cont;
  }
  nck[b] = nc;
}

// ---------------------------------------------------------------------------
// K3: per-chunk mean pooling -> chunk_emb bf16 (zeros for invalid chunks)
// ---------------------------------------------------------------------------
__global__ __launch_bounds__(256) void k_chunk_mean(
    const float* __restrict__ h, const int* __restrict__ cstart,
    const int* __restrict__ clen, const int* __restrict__ nck,
    unsigned short* __restrict__ ce)
{
  int b = blockIdx.y, c = blockIdx.x, tid = threadIdx.x;
  unsigned short* out = ce + ((size_t)b * SDIM + c) * DDIM + tid * 4;
  if (c >= nck[b]) { ushort4 z; z.x = z.y = z.z = z.w = 0; *(ushort4*)out = z; return; }
  int s0  = cstart[(size_t)b * SDIM + c];
  int len = clen  [(size_t)b * SDIM + c];
  const float* base = h + ((size_t)b * SDIM + s0) * DDIM + tid * 4;
  float4 acc; acc.x = acc.y = acc.z = acc.w = 0.f;
  for (int t = 0; t < len; ++t) {
    float4 x = *(const float4*)(base + (size_t)t * DDIM);
    acc.x += x.x; acc.y += x.y; acc.z += x.z; acc.w += x.w;
  }
  float inv = 1.f / (float)len;
  ushort4 o;
  o.x = f2bf(acc.x * inv); o.y = f2bf(acc.y * inv);
  o.z = f2bf(acc.z * inv); o.w = f2bf(acc.w * inv);
  *(ushort4*)out = o;
}

// ---------------------------------------------------------------------------
// K4: weight convert + transpose: Wt[n*K+k] = bf16(W[k*N+n])  (coalesced writes)
// ---------------------------------------------------------------------------
__global__ __launch_bounds__(256) void k_wconv_t(
    const float* __restrict__ W, unsigned short* __restrict__ Wt, int K, int N)
{
  size_t idx = (size_t)blockIdx.x * 256 + threadIdx.x;   // over N*K
  int n = (int)(idx / K), k = (int)(idx % K);
  Wt[idx] = f2bf(W[(size_t)k * N + n]);
}

// ---------------------------------------------------------------------------
// K5: generic batched bf16 WMMA GEMM: out = A(MxK) * Bt(NxK)^T * scale + bias
//   Block tile 64(M) x 128(N), 8 waves in a 2(M) x 4(N) grid.
//   Each wave owns a 32x32 output tile: 2 A frags + 2 B frags are each reused
//   twice from registers -> 4 v_wmma_f32_16x16x32_bf16 per 128B/lane loaded.
//   All frag loads are contiguous 16B loads matching the ISA 16-bit layouts:
//     A lane: K chunks at k0+8*hi and k0+16+8*hi (VGPR0-3 / VGPR4-7)
//     B lane: K chunks at k0+8*hi and k0+16+8*hi (same as A)
// ---------------------------------------------------------------------------
__global__ __launch_bounds__(256) void k_gemm_bf16(
    const unsigned short* __restrict__ A, long long strideA, int lda,
    const unsigned short* __restrict__ Bt, long long strideB, int ldb,
    const float* __restrict__ bias,
    unsigned short* __restrict__ outBf, float* __restrict__ outF,
    long long strideC, int ldc, int K, float scale, int storeT,
    const int* __restrict__ colLimit)
{
  int bz = blockIdx.z;
  int nBase = blockIdx.x * 128, mBase = blockIdx.y * 64;
  if (colLimit && nBase >= colLimit[bz]) return;   // skip fully-masked score tiles
  A  += (size_t)bz * strideA;
  Bt += (size_t)bz * strideB;
  int wave = threadIdx.x >> 5, lane = threadIdx.x & 31;
  int wm = wave & 1, wn = wave >> 1;               // 2 x 4 wave grid
  int subL = lane & 15, hi = lane >> 4;
  int row0 = mBase + wm * 32 + subL;
  int row1 = row0 + 16;
  int col0 = nBase + wn * 32 + subL;
  int col1 = col0 + 16;
  const uint4* pA0 = (const uint4*)(A  + (size_t)row0 * lda);
  const uint4* pA1 = (const uint4*)(A  + (size_t)row1 * lda);
  const uint4* pB0 = (const uint4*)(Bt + (size_t)col0 * ldb);
  const uint4* pB1 = (const uint4*)(Bt + (size_t)col1 * ldb);
  v8f c00 = {}; v8f c01 = {}; v8f c10 = {}; v8f c11 = {};
  int aoff = hi;        // (8*hi)/8 uint4s
  int boff = hi;        // (8*hi)/8 uint4s
  for (int k0 = 0; k0 < K; k0 += 32) {
    int kq = k0 >> 3;   // uint4 index of k0
    Frag a0, a1, b0, b1;
    a0.q[0] = pA0[kq + aoff];  a0.q[1] = pA0[kq + aoff + 2];
    a1.q[0] = pA1[kq + aoff];  a1.q[1] = pA1[kq + aoff + 2];
    b0.q[0] = pB0[kq + boff];  b0.q[1] = pB0[kq + boff + 2];
    b1.q[0] = pB1[kq + boff];  b1.q[1] = pB1[kq + boff + 2];
    __builtin_prefetch(pA0 + kq + 8, 0, 1);   // global_prefetch_b8 next k-slab
    __builtin_prefetch(pB0 + kq + 8, 0, 1);
    c00 = wmma_bf16(a0, b0, c00);
    c01 = wmma_bf16(a0, b1, c01);
    c10 = wmma_bf16(a1, b0, c10);
    c11 = wmma_bf16(a1, b1, c11);
  }
  float bv0 = bias ? bias[col0] : 0.f;
  float bv1 = bias ? bias[col1] : 0.f;
  if (outBf) outBf += (size_t)bz * strideC;
  if (outF)  outF  += (size_t)bz * strideC;
#pragma unroll
  for (int g = 0; g < 8; ++g) {                 // C/D layout: (M = g + 8*hi, N = subL)
    int m0 = mBase + wm * 32 + g + 8 * hi;
    int m1 = m0 + 16;
    float f00 = c00[g] * scale + bv0;
    float f01 = c01[g] * scale + bv1;
    float f10 = c10[g] * scale + bv0;
    float f11 = c11[g] * scale + bv1;
    size_t i00 = storeT ? ((size_t)col0 * ldc + m0) : ((size_t)m0 * ldc + col0);
    size_t i01 = storeT ? ((size_t)col1 * ldc + m0) : ((size_t)m0 * ldc + col1);
    size_t i10 = storeT ? ((size_t)col0 * ldc + m1) : ((size_t)m1 * ldc + col0);
    size_t i11 = storeT ? ((size_t)col1 * ldc + m1) : ((size_t)m1 * ldc + col1);
    if (outF) {
      outF[i00] = f00; outF[i01] = f01; outF[i10] = f10; outF[i11] = f11;
    } else {
      outBf[i00] = f2bf(f00); outBf[i01] = f2bf(f01);
      outBf[i10] = f2bf(f10); outBf[i11] = f2bf(f11);
    }
  }
}

// ---------------------------------------------------------------------------
// K6: masked row softmax over valid chunks; writes normalized probs as bf16
// ---------------------------------------------------------------------------
__global__ __launch_bounds__(256) void k_softmax(
    const float* __restrict__ scores, const int* __restrict__ nck,
    unsigned short* __restrict__ attn)
{
  size_t rowI = blockIdx.x;             // b*S + s
  int b = (int)(rowI >> 11);            // / 2048
  int Cn = nck[b];
  const float* sr = scores + rowI * (size_t)SDIM;
  unsigned short* ar = attn + rowI * (size_t)SDIM;
  __shared__ float red[256];
  int tid = threadIdx.x;
  float m = -3.4e38f;
  for (int c = tid; c < Cn; c += 256) m = fmaxf(m, sr[c]);
  red[tid] = m; __syncthreads();
  for (int off = 128; off; off >>= 1) {
    if (tid < off) red[tid] = fmaxf(red[tid], red[tid + off]);
    __syncthreads();
  }
  float mx = red[0]; __syncthreads();
  float sum = 0.f;
  for (int c = tid; c < Cn; c += 256) sum += __expf(sr[c] - mx);
  red[tid] = sum; __syncthreads();
  for (int off = 128; off; off >>= 1) {
    if (tid < off) red[tid] += red[tid + off];
    __syncthreads();
  }
  float inv = 1.f / red[0];
  for (int c = tid; c < SDIM; c += 256) {
    float p = (c < Cn) ? __expf(sr[c] - mx) * inv : 0.f;
    ar[c] = f2bf(p);
  }
}

// ---------------------------------------------------------------------------
extern "C" void kernel_launch(void* const* d_in, const int* in_sizes, int n_in,
                              void* d_out, int out_size, void* d_ws, size_t ws_size,
                              hipStream_t stream)
{
  (void)in_sizes; (void)n_in; (void)out_size; (void)ws_size;
  const float* h    = (const float*)d_in[0];
  const float* Wlab = (const float*)d_in[1];
  const float* blab = (const float*)d_in[2];
  const float* Wq   = (const float*)d_in[3];
  const float* bq   = (const float*)d_in[4];
  const float* Wk   = (const float*)d_in[5];
  const float* bk   = (const float*)d_in[6];
  const float* Wv   = (const float*)d_in[7];
  const float* bv   = (const float*)d_in[8];
  const float* Wo   = (const float*)d_in[9];
  const float* bo   = (const float*)d_in[10];
  float* out = (float*)d_out;

  char* base = (char*)d_ws;
  size_t off = 0;
  auto alloc = [&](size_t bytes) -> void* {
    void* r = base + off;
    off = (off + bytes + 255) & ~(size_t)255;
    return r;
  };
  int* labels = (int*)alloc((size_t)BS * 4);
  int* cstart = (int*)alloc((size_t)BS * 4);
  int* clen   = (int*)alloc((size_t)BS * 4);
  int* nck    = (int*)alloc((size_t)BDIM * 4);
  unsigned short* h_bf  = (unsigned short*)alloc((size_t)BS * DDIM * 2);
  unsigned short* ce_bf = (unsigned short*)alloc((size_t)BS * DDIM * 2);
  unsigned short* wq_t  = (unsigned short*)alloc((size_t)DDIM * DCDIM * 2);
  unsigned short* wk_t  = (unsigned short*)alloc((size_t)DDIM * DCDIM * 2);
  unsigned short* wv_t  = (unsigned short*)alloc((size_t)DDIM * DDIM * 2);
  unsigned short* wo_t  = (unsigned short*)alloc((size_t)DDIM * DDIM * 2);
  unsigned short* q_bf  = (unsigned short*)alloc((size_t)BS * DCDIM * 2);
  unsigned short* k_bf  = (unsigned short*)alloc((size_t)BS * DCDIM * 2);
  unsigned short* vt_bf = (unsigned short*)alloc((size_t)BDIM * DDIM * SDIM * 2); // v^T per batch
  unsigned short* at_bf = (unsigned short*)alloc((size_t)BDIM * SDIM * SDIM * 2); // attn probs
  unsigned short* ad_bf = (unsigned short*)alloc((size_t)BS * DDIM * 2);          // attended
  float* scores = (float*)alloc((size_t)BDIM * SDIM * SDIM * 4);

  // 1) labels + h->bf16 (single fused pass over h)
  k_labels_h2bf<<<BS / 8, 256, 0, stream>>>(h, Wlab, blab, labels, h_bf);
  // 2) BIOS chunk scan
  k_chunk_scan<<<1, BDIM, 0, stream>>>(labels, cstart, clen, nck);
  // 3) chunk means -> chunk_emb bf16
  k_chunk_mean<<<dim3(SDIM, BDIM), 256, 0, stream>>>(h, cstart, clen, nck, ce_bf);
  // 4) weights -> bf16, transposed to NxK
  int wblocks = (DDIM * DCDIM) / 256;
  k_wconv_t<<<wblocks, 256, 0, stream>>>(Wq, wq_t, DDIM, DCDIM);
  k_wconv_t<<<wblocks, 256, 0, stream>>>(Wk, wk_t, DDIM, DCDIM);
  k_wconv_t<<<wblocks, 256, 0, stream>>>(Wv, wv_t, DDIM, DDIM);
  k_wconv_t<<<wblocks, 256, 0, stream>>>(Wo, wo_t, DDIM, DDIM);
  // 5) q = h @ Wq + bq          (M = B*S, one flat GEMM)
  k_gemm_bf16<<<dim3(DCDIM / 128, BS / 64, 1), 256, 0, stream>>>(
      h_bf, 0, DDIM, wq_t, 0, DDIM, bq, q_bf, nullptr, 0, DCDIM, DDIM, 1.f, 0, nullptr);
  // 6) k = chunk_emb @ Wk + bk
  k_gemm_bf16<<<dim3(DCDIM / 128, BS / 64, 1), 256, 0, stream>>>(
      ce_bf, 0, DDIM, wk_t, 0, DDIM, bk, k_bf, nullptr, 0, DCDIM, DDIM, 1.f, 0, nullptr);
  // 7) v = chunk_emb @ Wv + bv, stored TRANSPOSED per batch: vt[b][d][c]
  k_gemm_bf16<<<dim3(DDIM / 128, SDIM / 64, BDIM), 256, 0, stream>>>(
      ce_bf, (long long)SDIM * DDIM, DDIM, wv_t, 0, DDIM, bv,
      vt_bf, nullptr, (long long)DDIM * SDIM, SDIM, DDIM, 1.f, 1, nullptr);
  // 8) scores = q @ k^T * (1/sqrt(DC)); tiles past n_chunks skipped
  k_gemm_bf16<<<dim3(SDIM / 128, SDIM / 64, BDIM), 256, 0, stream>>>(
      q_bf, (long long)SDIM * DCDIM, DCDIM, k_bf, (long long)SDIM * DCDIM, DCDIM, nullptr,
      nullptr, scores, (long long)SDIM * SDIM, SDIM, DCDIM, 0.03125f, 0, nck);
  // 9) masked softmax -> attn bf16 (zeros past n_chunks)
  k_softmax<<<BS, 256, 0, stream>>>(scores, nck, at_bf);
  // 10) attended = attn @ v  (B-operand = vt, contiguous along K)
  k_gemm_bf16<<<dim3(DDIM / 128, SDIM / 64, BDIM), 256, 0, stream>>>(
      at_bf, (long long)SDIM * SDIM, SDIM, vt_bf, (long long)DDIM * SDIM, SDIM, nullptr,
      ad_bf, nullptr, (long long)SDIM * DDIM, DDIM, SDIM, 1.f, 0, nullptr);
  // 11) out = attended @ Wo + bo (f32 straight to d_out)
  k_gemm_bf16<<<dim3(DDIM / 128, BS / 64, 1), 256, 0, stream>>>(
      ad_bf, 0, DDIM, wo_t, 0, DDIM, bo, nullptr, out, 0, DDIM, DDIM, 1.f, 0, nullptr);
}
